// RBFLayer_56461640073237
// MI455X (gfx1250) — matrix operand order repaired and checked by reading, not compiled, not run on hardware
//
#include <hip/hip_runtime.h>

// ---------------------------------------------------------------------------
// RBF layer: out[b,k] = exp(-beta[k] * max(||x_b||^2 + ||c_k||^2 - 2*<x_b,c_k>, 0))
// B=16384, K=4096, D=1024 (fp32 in/out).
// Pass 1 (prep): fp32 -> bf16 copies of x/centers into d_ws + row sum-of-squares.
// Pass 2 (GEMM): bf16 WMMA 16x16x32, f32 accum, BK=64 K-chunks, double-buffered
//                GLOBAL_LOAD_ASYNC_TO_LDS_B128 staging, exp epilogue.
// Workspace requirement: (B+K)*4 + (B+K)*D*2 bytes  (~40.1 MB for given shapes).
// ---------------------------------------------------------------------------

typedef __attribute__((ext_vector_type(16))) __bf16 v16bf;
typedef __attribute__((ext_vector_type(8)))  float  v8f;

union FragAB {
    v16bf v;
    uint4 q[2];
};

__device__ __forceinline__ unsigned pack_bf16(float a, float b) {
    unsigned ua = __float_as_uint(a);
    unsigned ub = __float_as_uint(b);
    ua = (ua + 0x7fffu + ((ua >> 16) & 1u)) >> 16;   // round-to-nearest-even
    ub = (ub + 0x7fffu + ((ub >> 16) & 1u)) >> 16;
    return ua | (ub << 16);
}

// ---------------------------------------------------------------------------
// Prep: per row, write bf16 copy and row sum-of-squares. One block per row.
// ---------------------------------------------------------------------------
__global__ void rbf_prep(const float* __restrict__ X,
                         unsigned short* __restrict__ Xb,
                         float* __restrict__ nrm, int D) {
    const int row = blockIdx.x;
    const float4* p = reinterpret_cast<const float4*>(X + (size_t)row * D);
    uint2* q = reinterpret_cast<uint2*>(Xb + (size_t)row * D);
    float s = 0.0f;
    for (int i = threadIdx.x; i < (D >> 2); i += blockDim.x) {
        float4 v = p[i];
        s += v.x * v.x + v.y * v.y + v.z * v.z + v.w * v.w;
        q[i] = make_uint2(pack_bf16(v.x, v.y), pack_bf16(v.z, v.w));
    }
    for (int off = 16; off > 0; off >>= 1)
        s += __shfl_down(s, off, 32);
    __shared__ float red[8];
    const int lane = threadIdx.x & 31;
    const int wv   = threadIdx.x >> 5;
    if (lane == 0) red[wv] = s;
    __syncthreads();
    if (threadIdx.x == 0) {
        float t = 0.0f;
        for (int w = 0; w < 8; ++w) t += red[w];
        nrm[row] = t;
    }
}

// ---------------------------------------------------------------------------
// Main tiled WMMA kernel.
// Block tile: 128x128, 256 threads = 8 waves in 4(M) x 2(N) grid.
// Each wave: 32x64 = 2x4 WMMA tiles; BK=64 -> 16 WMMAs per barrier pair.
// Double-buffered LDS fed by GLOBAL_LOAD_ASYNC_TO_LDS_B128 (8 copies/wave/buf).
// ---------------------------------------------------------------------------
#define LDS_STRIDE 72                       // halves per row (128B data + 16B pad)
#define ROW_BYTES  (LDS_STRIDE * 2)         // 144 B = 36 dwords
#define TILE_BYTES (128 * ROW_BYTES)        // one buffer: 18432 B

__global__ void __launch_bounds__(256)
rbf_wmma_kernel(const unsigned short* __restrict__ Xb,
                const unsigned short* __restrict__ Cb,
                const float* __restrict__ betas,
                const float* __restrict__ x2g,
                const float* __restrict__ c2g,
                float* __restrict__ out,
                int Kn, int Dn) {
    __shared__ __align__(16) unsigned short sA[2][128 * LDS_STRIDE];
    __shared__ __align__(16) unsigned short sB[2][128 * LDS_STRIDE];

    const int tid  = threadIdx.x;
    const int lane = tid & 31;
    const int wv   = tid >> 5;
    const int waveM = wv >> 1;          // 0..3
    const int waveN = wv & 1;           // 0..1
    const int m0 = waveM * 32;
    const int n0 = waveN * 64;

    const int bN = blockIdx.x * 128;    // center (output col) base
    const int bM = blockIdx.y * 128;    // x row (output row) base

    const int half = lane >> 4;
    const int l15  = lane & 15;

    v8f acc[2][4];
#pragma unroll
    for (int mt = 0; mt < 2; ++mt)
#pragma unroll
        for (int nt = 0; nt < 4; ++nt)
#pragma unroll
            for (int e = 0; e < 8; ++e)
                acc[mt][nt][e] = 0.0f;

    // ---- async staging setup -------------------------------------------------
    // Tile = 128 rows x 64 bf16 = 1024 16B-chunks. chunk = wv*128 + j*32 + lane,
    // j = 0..3. Each wave issues 4 b128 async copies for A and 4 for B per buf.
    unsigned offc[4];
    unsigned long long gA[4], gB[4];
#pragma unroll
    for (int j = 0; j < 4; ++j) {
        const unsigned chunk = (unsigned)(wv * 128 + j * 32 + lane);
        const unsigned row = chunk >> 3, cw = chunk & 7;
        offc[j] = row * ROW_BYTES + cw * 16;
        gA[j] = (unsigned long long)(size_t)Xb +
                (((size_t)(bM + row) * Dn + cw * 8) << 1);
        gB[j] = (unsigned long long)(size_t)Cb +
                (((size_t)(bN + row) * Dn + cw * 8) << 1);
    }
    const unsigned ldsA0 = (unsigned)(size_t)&sA[0][0];
    const unsigned ldsB0 = (unsigned)(size_t)&sB[0][0];

#define ISSUE_TILE(buf, d0)                                                     \
    do {                                                                        \
        const unsigned long long koff = (unsigned long long)(d0) << 1;          \
        _Pragma("unroll")                                                       \
        for (int j = 0; j < 4; ++j)                                             \
            asm volatile("global_load_async_to_lds_b128 %0, %1, off"            \
                         :: "v"(ldsA0 + (buf) * TILE_BYTES + offc[j]),          \
                            "v"(gA[j] + koff) : "memory");                      \
        _Pragma("unroll")                                                       \
        for (int j = 0; j < 4; ++j)                                             \
            asm volatile("global_load_async_to_lds_b128 %0, %1, off"            \
                         :: "v"(ldsB0 + (buf) * TILE_BYTES + offc[j]),          \
                            "v"(gB[j] + koff) : "memory");                      \
    } while (0)

    // 16 WMMAs on one LDS buffer (two K=32 sub-steps).
#define COMPUTE_TILE(buf)                                                       \
    do {                                                                        \
        _Pragma("unroll")                                                       \
        for (int ks = 0; ks < 2; ++ks) {                                        \
            FragAB afr[2];                                                      \
            _Pragma("unroll")                                                   \
            for (int mt = 0; mt < 2; ++mt) {                                    \
                const int base =                                                \
                    (m0 + mt * 16 + l15) * LDS_STRIDE + ks * 32 + half * 8;     \
                afr[mt].q[0] = *reinterpret_cast<const uint4*>(&sA[buf][base]); \
                afr[mt].q[1] =                                                  \
                    *reinterpret_cast<const uint4*>(&sA[buf][base + 16]);       \
            }                                                                   \
            FragAB bfr[4];                                                      \
            _Pragma("unroll")                                                   \
            for (int nt = 0; nt < 4; ++nt) {                                    \
                const int base =                                                \
                    (n0 + nt * 16 + l15) * LDS_STRIDE + ks * 32 + half * 8;     \
                bfr[nt].q[0] = *reinterpret_cast<const uint4*>(&sB[buf][base]); \
                bfr[nt].q[1] =                                                  \
                    *reinterpret_cast<const uint4*>(&sB[buf][base + 16]);       \
            }                                                                   \
            _Pragma("unroll")                                                   \
            for (int mt = 0; mt < 2; ++mt)                                      \
                _Pragma("unroll")                                               \
                for (int nt = 0; nt < 4; ++nt)                                  \
                    acc[mt][nt] = __builtin_amdgcn_wmma_f32_16x16x32_bf16(      \
                        false, afr[mt].v, false, bfr[nt].v,                     \
                        (short)0, acc[mt][nt], false, false);                   \
        }                                                                       \
    } while (0)

    const int niter = Dn >> 6;          // D / 64

    ISSUE_TILE(0, 0);

    // steady state: issue next buffer, retire current (in-order per wave), sync.
    for (int it = 0; it < niter - 1; ++it) {
        const int buf = it & 1;
        ISSUE_TILE(buf ^ 1, (it + 1) * 64);
        asm volatile("s_wait_asynccnt 8" ::: "memory");
        __syncthreads();                // `buf` fully staged by all waves
        COMPUTE_TILE(buf);
        __syncthreads();                // all done reading `buf` before refill
    }
    // peeled last iteration: nothing left to issue.
    asm volatile("s_wait_asynccnt 0" ::: "memory");
    __syncthreads();
    COMPUTE_TILE((niter - 1) & 1);

    // ---- epilogue: d2 = max(x2 + c2 - 2*cross, 0); out = exp(-beta*d2) ----
    // C/D layout: VGPR e, lanes 0-15 -> M = e, N = lane; lanes 16-31 -> M = 8+e.
    float x2v[2][8];
#pragma unroll
    for (int mt = 0; mt < 2; ++mt)
#pragma unroll
        for (int e = 0; e < 8; ++e)
            x2v[mt][e] = x2g[bM + m0 + mt * 16 + half * 8 + e];

    float c2v[4], bev[4];
#pragma unroll
    for (int nt = 0; nt < 4; ++nt) {
        const int gN = bN + n0 + nt * 16 + l15;
        c2v[nt] = c2g[gN];
        bev[nt] = betas[gN];
    }

#pragma unroll
    for (int mt = 0; mt < 2; ++mt) {
#pragma unroll
        for (int nt = 0; nt < 4; ++nt) {
            const int gN = bN + n0 + nt * 16 + l15;
#pragma unroll
            for (int e = 0; e < 8; ++e) {
                const int gM = bM + m0 + mt * 16 + half * 8 + e;
                float cross = acc[mt][nt][e];
                float d2 = fmaxf(x2v[mt][e] + c2v[nt] - 2.0f * cross, 0.0f);
                out[(size_t)gM * Kn + gN] = __expf(-bev[nt] * d2);
            }
        }
    }
#undef ISSUE_TILE
#undef COMPUTE_TILE
}

// ---------------------------------------------------------------------------
// Host launcher. d_in = {x (B,D) f32, centers (K,D) f32, betas (K,) f32}.
// d_out = (B,K) f32. d_ws: x2[B] f32 | c2[K] f32 | Xbf[B*D] bf16 | Cbf[K*D] bf16.
// ---------------------------------------------------------------------------
extern "C" void kernel_launch(void* const* d_in, const int* in_sizes, int n_in,
                              void* d_out, int out_size, void* d_ws, size_t ws_size,
                              hipStream_t stream) {
    const float* x       = (const float*)d_in[0];
    const float* centers = (const float*)d_in[1];
    const float* betas   = (const float*)d_in[2];
    float* out           = (float*)d_out;

    const int K = in_sizes[2];             // 4096
    const int D = in_sizes[1] / K;         // 1024
    const int B = in_sizes[0] / D;         // 16384

    float* x2 = (float*)d_ws;
    float* c2 = x2 + B;
    unsigned short* Xb = (unsigned short*)(c2 + K);
    unsigned short* Cb = Xb + (size_t)B * D;

    rbf_prep<<<B, 256, 0, stream>>>(x, Xb, x2, D);
    rbf_prep<<<K, 256, 0, stream>>>(centers, Cb, c2, D);

    dim3 grid(K / 128, B / 128);
    rbf_wmma_kernel<<<grid, 256, 0, stream>>>(Xb, Cb, betas, x2, c2, out, K, D);
}
